// MultiQueryAttention_18743237279847
// MI455X (gfx1250) — matrix operand order repaired and checked by reading, not compiled, not run on hardware
//
#include <hip/hip_runtime.h>

// ---------------------------------------------------------------------------
// MQA forward for MI455X (gfx1250, wave32, WMMA bf16 w/ f32 accumulate)
// B=2, S=2048, E=2048, H=16, D=128, QKV width = (H+2)*D = 2304
// ---------------------------------------------------------------------------

typedef __attribute__((ext_vector_type(16))) __bf16 v16bf;
typedef __attribute__((ext_vector_type(8)))  __bf16 v8bf;
typedef __attribute__((ext_vector_type(4)))  __bf16 v4bf;
typedef __attribute__((ext_vector_type(8)))  float  v8f;
typedef __attribute__((ext_vector_type(4)))  int    v4i;

#define AS1 __attribute__((address_space(1)))
#define AS3 __attribute__((address_space(3)))

#if __has_builtin(__builtin_amdgcn_global_load_async_to_lds_b128)
#define HAVE_ASYNC 1
#endif

__device__ __forceinline__ v8bf ld8(const __bf16* p) {
    return *reinterpret_cast<const v8bf*>(p);
}

__device__ __forceinline__ v16bf make_frag(v8bf lo, v8bf hi) {
    v16bf r;
#pragma unroll
    for (int i = 0; i < 8; ++i) { r[i] = lo[i]; r[i + 8] = hi[i]; }
    return r;
}

__device__ __forceinline__ v8f wmma_bf16(v16bf a, v16bf b, v8f c) {
    // D = A(16x32 bf16) * B(32x16 bf16) + C(16x16 f32)
    return __builtin_amdgcn_wmma_f32_16x16x32_bf16(
        false, a, false, b, (short)0, c, false, false);
}

#ifdef HAVE_ASYNC
__device__ __forceinline__ void wait_async0() {
#if __has_builtin(__builtin_amdgcn_s_wait_asynccnt)
    __builtin_amdgcn_s_wait_asynccnt(0);
#else
    asm volatile("s_wait_asynccnt 0x0" ::: "memory");
#endif
}
__device__ __forceinline__ void wait_async1() {
#if __has_builtin(__builtin_amdgcn_s_wait_asynccnt)
    __builtin_amdgcn_s_wait_asynccnt(1);
#else
    asm volatile("s_wait_asynccnt 0x1" ::: "memory");
#endif
}
#endif

// ---------------------------------------------------------------------------
// fp32 -> bf16 convert (n multiple of 1024; 4 elems/thread)
// ---------------------------------------------------------------------------
__global__ __launch_bounds__(256) void f32_to_bf16_kernel(
    const float* __restrict__ in, __bf16* __restrict__ out, int n) {
    int i = (blockIdx.x * 256 + threadIdx.x) * 4;
    if (i + 3 >= n) return;
    float4 v = *reinterpret_cast<const float4*>(in + i);
    v4bf o;
    o[0] = (__bf16)v.x; o[1] = (__bf16)v.y; o[2] = (__bf16)v.z; o[3] = (__bf16)v.w;
    *reinterpret_cast<v4bf*>(out + i) = o;
}

// ---------------------------------------------------------------------------
// Tiled transpose + convert: in fp32 [R][C] -> out bf16 [C][R] (R,C mult of 32)
// ---------------------------------------------------------------------------
__global__ __launch_bounds__(256) void transpose_f32_to_bf16(
    const float* __restrict__ in, __bf16* __restrict__ out, int R, int C) {
    __shared__ float tile[32][33];
    const int r0 = blockIdx.y * 32, c0 = blockIdx.x * 32;
    const int c = threadIdx.x & 31, r = threadIdx.x >> 5;   // r = 0..7
#pragma unroll
    for (int i = 0; i < 4; ++i)
        tile[r + 8 * i][c] = in[(size_t)(r0 + r + 8 * i) * C + c0 + c];
    __syncthreads();
#pragma unroll
    for (int i = 0; i < 4; ++i)
        out[(size_t)(c0 + r + 8 * i) * R + r0 + c] = (__bf16)tile[c][r + 8 * i];
}

// ---------------------------------------------------------------------------
// V transpose: qkv bf16 [B*S][2304] (v at col 2176) -> vT [B][128][S]
// ---------------------------------------------------------------------------
__global__ __launch_bounds__(256) void transpose_v_kernel(
    const __bf16* __restrict__ qkv, __bf16* __restrict__ vT) {
    __shared__ __bf16 tile[32][33];
    const int b = blockIdx.z;
    const int t0 = blockIdx.x * 32, d0 = blockIdx.y * 32;
    const int c = threadIdx.x & 31, r = threadIdx.x >> 5;
#pragma unroll
    for (int i = 0; i < 4; ++i)
        tile[r + 8 * i][c] =
            qkv[(size_t)(b * 2048 + t0 + r + 8 * i) * 2304 + 2176 + d0 + c];
    __syncthreads();
#pragma unroll
    for (int i = 0; i < 4; ++i)
        vT[((size_t)b * 128 + d0 + r + 8 * i) * 2048 + t0 + c] = tile[c][r + 8 * i];
}

// ---------------------------------------------------------------------------
// bf16 GEMM: C[M,N] = A[M,K] * BT[N,K]^T.  Block 128x128, 8 waves of 32x64.
// A tile double-buffered in LDS via async-to-LDS copies; B fragments are
// two 16-byte runs per lane (k 8h..8h+7, 16+8h..16+8h+7) loaded directly from global (BT row-major).
// ---------------------------------------------------------------------------
#define LDPAD 40

template <bool OUT_BF16>
__global__ __launch_bounds__(256) void gemm_bt_wmma(
    const __bf16* __restrict__ A, const __bf16* __restrict__ BT,
    float* __restrict__ Cf, __bf16* __restrict__ Cb,
    int M, int N, int K) {
    __shared__ __bf16 As[2][128 * LDPAD];

    const int t    = threadIdx.x;
    const int lane = t & 31;
    const int wave = t >> 5;
    const int lr   = lane & 15;
    const int hs   = lane >> 4;
    const int m0   = blockIdx.y * 128;
    const int n0   = blockIdx.x * 128;
    const int wm   = (wave >> 1) * 32;   // 4 waves along M
    const int wn   = (wave & 1) * 64;    // 2 waves along N

    v8f acc[2][4] = {};
    const int arow = t >> 1, ahc = (t & 1) * 16;   // A staging coords: 16B/thread

    auto issueA = [&](int k0, int buf) {
        const __bf16* gp = A + (size_t)(m0 + arow) * K + k0 + ahc;
        __bf16* lp = &As[buf][arow * LDPAD + ahc];
#ifdef HAVE_ASYNC
        __builtin_amdgcn_global_load_async_to_lds_b128(
            (AS1 v4i*)gp, (AS3 v4i*)lp, 0, 0);
#else
        *reinterpret_cast<v8bf*>(lp) = ld8(gp);
#endif
    };

    const int KT = K >> 5;
    issueA(0, 0);
    for (int kt = 0; kt < KT; ++kt) {
        const int buf = kt & 1;
        const int k0  = kt << 5;
        const bool more = (kt + 1 < KT);
        if (more) issueA(k0 + 32, buf ^ 1);
#ifdef HAVE_ASYNC
        if (more) wait_async1(); else wait_async0();
#endif
        __syncthreads();   // A tile[buf] visible to all waves

        // B fragments straight from global BT (L2-resident weights)
        v16bf af[2], bfr[4];
#pragma unroll
        for (int j = 0; j < 4; ++j) {
            const __bf16* bp = BT + (size_t)(n0 + wn + j * 16 + lr) * K + k0 + hs * 8;
            bfr[j] = make_frag(ld8(bp), ld8(bp + 16));
            if (more) __builtin_prefetch(bp + 32, 0, 0);
        }
#pragma unroll
        for (int i = 0; i < 2; ++i) {
            const __bf16* p = &As[buf][(wm + i * 16 + lr) * LDPAD + hs * 8];
            af[i] = make_frag(ld8(p), ld8(p + 16));   // K 0..15 / 16..31 split
        }
#pragma unroll
        for (int i = 0; i < 2; ++i)
#pragma unroll
            for (int j = 0; j < 4; ++j)
                acc[i][j] = wmma_bf16(af[i], bfr[j], acc[i][j]);
        __syncthreads();   // all waves done reading tile[buf] before overwrite
    }

    // epilogue: C layout = lane(l&15)=col, vgpr r => row r + 8*(l>>4)
#pragma unroll
    for (int i = 0; i < 2; ++i)
#pragma unroll
        for (int j = 0; j < 4; ++j) {
            const int col = n0 + wn + j * 16 + lr;
#pragma unroll
            for (int r = 0; r < 8; ++r) {
                const int row = m0 + wm + i * 16 + hs * 8 + r;
                if (OUT_BF16) Cb[(size_t)row * N + col] = (__bf16)acc[i][j][r];
                else          Cf[(size_t)row * N + col] = acc[i][j][r];
            }
        }
}

// ---------------------------------------------------------------------------
// Flash MQA attention, barrier-free inner loop.
// qkv: [B*S][2304] bf16 (q at h*128, k at 2048);  vT: [B][128][S] bf16.
// Grid (S/128, H, B); wave w owns 16 q rows; loops over 32-key blocks.
// ---------------------------------------------------------------------------
__global__ __launch_bounds__(256) void mqa_flash_kernel(
    const __bf16* __restrict__ qkv, const __bf16* __restrict__ vT,
    __bf16* __restrict__ attn) {
    constexpr int S = 2048, QKVN = 2304, E = 2048;
    __shared__ __bf16 pS[8][16 * LDPAD];     // per-wave P[row][key]

    const int t = threadIdx.x, lane = t & 31, wave = t >> 5;
    const int lr = lane & 15, hs = lane >> 4;
    const int b = blockIdx.z, h = blockIdx.y, qb = blockIdx.x;
    const int qr0 = qb * 128 + wave * 16;

    // Q fragments: 4 K=32 slices covering D=128 (resident in VGPRs)
    const size_t rowq = (size_t)(b * S + qr0 + lr) * QKVN + h * 128;
    v16bf qf[4];
#pragma unroll
    for (int s = 0; s < 4; ++s) {
        const __bf16* p = qkv + rowq + s * 32 + hs * 8;
        qf[s] = make_frag(ld8(p), ld8(p + 16));
    }

    v8f o[8] = {};
    float m[8], lsum[8];
#pragma unroll
    for (int i = 0; i < 8; ++i) { m[i] = -1e30f; lsum[i] = 0.f; }
    const float smscale = 0.08838834764831845f;  // 1/sqrt(128)
    const size_t kvbase = (size_t)b * S * QKVN;
    const size_t vbase  = (size_t)b * 128 * S;

    for (int kb = 0; kb < S; kb += 32) {
        // ---- S = Q * K^T : K^T fragments contiguous per lane from global ----
        v8f s0 = {}, s1 = {};
#pragma unroll
        for (int s = 0; s < 4; ++s) {
            const __bf16* kp0 = qkv + kvbase + (size_t)(kb + lr) * QKVN      + 2048 + s * 32 + hs * 8;
            const __bf16* kp1 = qkv + kvbase + (size_t)(kb + 16 + lr) * QKVN + 2048 + s * 32 + hs * 8;
            v16bf bk0 = make_frag(ld8(kp0), ld8(kp0 + 16));
            v16bf bk1 = make_frag(ld8(kp1), ld8(kp1 + 16));
            s0 = wmma_bf16(qf[s], bk0, s0);
            s1 = wmma_bf16(qf[s], bk1, s1);
        }
        if (kb + 32 < S) {  // prefetch next key block (global_prefetch_b8)
            __builtin_prefetch(qkv + kvbase + (size_t)(kb + 32 + lr) * QKVN + 2048, 0, 0);
            __builtin_prefetch(vT + vbase + (size_t)lane * S + kb + 32, 0, 0);
        }

        // ---- online softmax (rows live across 16-lane halves) ----
#pragma unroll
        for (int i = 0; i < 8; ++i) {
            float a0 = s0[i] * smscale, a1 = s1[i] * smscale;
            float mx = fmaxf(a0, a1);
#pragma unroll
            for (int off = 1; off < 16; off <<= 1)
                mx = fmaxf(mx, __shfl_xor(mx, off, 32));
            float mn   = fmaxf(m[i], mx);
            float corr = __expf(m[i] - mn);
            m[i] = mn;
            float p0 = __expf(a0 - mn), p1 = __expf(a1 - mn);
            float rs = p0 + p1;
#pragma unroll
            for (int off = 1; off < 16; off <<= 1)
                rs += __shfl_xor(rs, off, 32);
            lsum[i] = lsum[i] * corr + rs;
#pragma unroll
            for (int j = 0; j < 8; ++j) o[j][i] *= corr;
            // spill P to per-wave LDS (C layout -> row-major)
            pS[wave][(i + hs * 8) * LDPAD + lr]      = (__bf16)p0;
            pS[wave][(i + hs * 8) * LDPAD + 16 + lr] = (__bf16)p1;
        }
        asm volatile("s_wait_dscnt 0x0" ::: "memory");

        // ---- O += P * V : V^T fragments contiguous per lane from global ----
        v16bf pf;
        {
            const __bf16* pp = &pS[wave][lr * LDPAD + hs * 8];
            pf = make_frag(ld8(pp), ld8(pp + 16));
        }
#pragma unroll
        for (int j = 0; j < 8; ++j) {
            const __bf16* vp = vT + vbase + (size_t)(j * 16 + lr) * S + kb + hs * 8;
            v16bf bv = make_frag(ld8(vp), ld8(vp + 16));
            o[j] = wmma_bf16(pf, bv, o[j]);
        }
    }

    // ---- normalize and write attn (layout [B,S,H*D]) ----
    float inv[8];
#pragma unroll
    for (int i = 0; i < 8; ++i) inv[i] = 1.0f / lsum[i];
#pragma unroll
    for (int j = 0; j < 8; ++j)
#pragma unroll
        for (int i = 0; i < 8; ++i) {
            const int row = qr0 + i + hs * 8;
            const int col = h * 128 + j * 16 + lr;
            attn[(size_t)(b * S + row) * E + col] = (__bf16)(o[j][i] * inv[i]);
        }
}

// ---------------------------------------------------------------------------
// Launch
// ---------------------------------------------------------------------------
extern "C" void kernel_launch(void* const* d_in, const int* in_sizes, int n_in,
                              void* d_out, int out_size, void* d_ws, size_t ws_size,
                              hipStream_t stream) {
    constexpr int B = 2, S = 2048, E = 2048, H = 16, D = 128;
    constexpr int QKVN = (H + 2) * D;         // 2304
    constexpr int MT = B * S;                 // 4096 rows

    const float* x      = (const float*)d_in[0];
    const float* w_attn = (const float*)d_in[1];
    const float* w_out  = (const float*)d_in[2];
    float*       out    = (float*)d_out;

    char* ws = (char*)d_ws;
    __bf16* x_bf    = (__bf16*)(ws);                       // 16 MiB  [MT][E]
    __bf16* waT     = (__bf16*)(ws + 16777216);            //  9 MiB  [QKVN][E]
    __bf16* woT     = (__bf16*)(ws + 26214400);            //  8 MiB  [E][E] (T)
    __bf16* qkv_bf  = (__bf16*)(ws + 34603008);            // 18 MiB  [MT][QKVN]
    __bf16* attn_bf = (__bf16*)(ws + 53477376);            // 16 MiB  [MT][E]
    __bf16* vT_bf   = (__bf16*)(ws + 70254592);            //  1 MiB  [B][D][S]

    const int nX = MT * E;
    f32_to_bf16_kernel<<<nX / 1024, 256, 0, stream>>>(x, x_bf, nX);
    // one-time weight transposes so GEMM B-fragments are contiguous per lane
    transpose_f32_to_bf16<<<dim3(QKVN / 32, E / 32), 256, 0, stream>>>(w_attn, waT, E, QKVN);
    transpose_f32_to_bf16<<<dim3(E / 32, E / 32), 256, 0, stream>>>(w_out, woT, E, E);

    // QKV = X * W_attn   (bf16 out)
    gemm_bt_wmma<true><<<dim3(QKVN / 128, MT / 128), 256, 0, stream>>>(
        x_bf, waT, nullptr, qkv_bf, MT, QKVN, E);

    // V^T for contiguous P*V fragments
    transpose_v_kernel<<<dim3(S / 32, D / 32, B), 256, 0, stream>>>(qkv_bf, vT_bf);

    // Flash MQA attention
    mqa_flash_kernel<<<dim3(S / 128, H, B), 256, 0, stream>>>(qkv_bf, vT_bf, attn_bf);

    // Out = Attn * W_out  (fp32 out)
    gemm_bt_wmma<false><<<dim3(E / 128, MT / 128), 256, 0, stream>>>(
        attn_bf, woT, out, nullptr, MT, E, E);
}
